// GPT2Block_23493471109954
// MI455X (gfx1250) — matrix operand order repaired and checked by reading, not compiled, not run on hardware
//
#include <hip/hip_runtime.h>
#include <hip/hip_bf16.h>

typedef __bf16 bf16_t;
typedef __attribute__((ext_vector_type(16))) __bf16 v16bf;
typedef __attribute__((ext_vector_type(8)))  float  v8f;

union Frag {
    v16bf   v;
    unsigned u[8];
    __bf16  h[16];
};

#define EMBD 1024
#define SEQ  2048
#define NB   2
#define HEADS 16
#define HD   64
#define FF   4096
#define MTOT (NB * SEQ)   // 4096 rows
#define QKVLD (3 * EMBD)  // 3072

#if __has_builtin(__builtin_amdgcn_global_load_async_to_lds_b128)
#define HAVE_ASYNC 1
typedef int v4i __attribute__((vector_size(16)));
#else
#define HAVE_ASYNC 0
#endif

// ---------------------------------------------------------------------------
// fp32 [K][N]  ->  bf16 [N][K]  (transpose + convert), LDS-tiled, coalesced
// grid(N/32, K/32), block(32,8)
// ---------------------------------------------------------------------------
__global__ __launch_bounds__(256) void transpose_cvt(const float* __restrict__ in,
                                                     bf16_t* __restrict__ out,
                                                     int K, int N) {
    __shared__ float tile[32][33];
    int bx = blockIdx.x * 32;
    int by = blockIdx.y * 32;
    int tx = threadIdx.x;   // 0..31
    int ty = threadIdx.y;   // 0..7
#pragma unroll
    for (int i = 0; i < 32; i += 8)
        tile[ty + i][tx] = in[(size_t)(by + ty + i) * N + bx + tx];
    __syncthreads();
#pragma unroll
    for (int i = 0; i < 32; i += 8)
        out[(size_t)(bx + ty + i) * K + by + tx] = (bf16_t)tile[tx][ty + i];
}

// ---------------------------------------------------------------------------
// LayerNorm over last dim (1024), one row per 256-thread block, bf16 output
// ---------------------------------------------------------------------------
__global__ __launch_bounds__(256) void ln_bf16(const float* __restrict__ x,
                                               const float* __restrict__ scale,
                                               const float* __restrict__ bias,
                                               bf16_t* __restrict__ out) {
    __shared__ float red[256];
    int row = blockIdx.x;
    int t = threadIdx.x;
    const float* xr = x + (size_t)row * EMBD;

    float v[4];
    float s = 0.f;
#pragma unroll
    for (int i = 0; i < 4; ++i) { v[i] = xr[t + 256 * i]; s += v[i]; }

    red[t] = s; __syncthreads();
    for (int off = 128; off > 0; off >>= 1) {
        if (t < off) red[t] += red[t + off];
        __syncthreads();
    }
    float mu = red[0] * (1.0f / EMBD);
    __syncthreads();

    float s2 = 0.f;
#pragma unroll
    for (int i = 0; i < 4; ++i) { float d = v[i] - mu; s2 += d * d; }
    red[t] = s2; __syncthreads();
    for (int off = 128; off > 0; off >>= 1) {
        if (t < off) red[t] += red[t + off];
        __syncthreads();
    }
    float rstd = rsqrtf(red[0] * (1.0f / EMBD) + 1e-5f);

    bf16_t* orow = out + (size_t)row * EMBD;
#pragma unroll
    for (int i = 0; i < 4; ++i) {
        int c = t + 256 * i;
        orow[c] = (bf16_t)((v[i] - mu) * rstd * scale[c] + bias[c]);
    }
}

// ---------------------------------------------------------------------------
// Tiled bf16 WMMA GEMM: C[M,N] = A[M,K] * Wt[N,K]^T + bias (+relu)(+residual)
// 256 threads = 8 waves; 128x128 block tile; K-step 64.
// Double-buffered LDS staging via async global->LDS DMA (ASYNCcnt):
// tile i+1 streams into the alternate buffer underneath tile i's 16 WMMAs.
// ---------------------------------------------------------------------------
#define BM 128
#define BN 128
#define BK 64

__global__ __launch_bounds__(256) void gemm_bf16_wmma(
    const bf16_t* __restrict__ A, const bf16_t* __restrict__ Wt,
    const float* __restrict__ bias, const float* __restrict__ residual,
    float* __restrict__ outF, bf16_t* __restrict__ outB,
    int N, int K, int relu) {

    __shared__ __align__(16) bf16_t As[2][BM][BK];   // [buf][m][k]
    __shared__ __align__(16) bf16_t Bs[2][BN][BK];   // [buf][n][k]

    int tid  = threadIdx.x;
    int bm   = blockIdx.y * BM;
    int bn   = blockIdx.x * BN;
    int wave = tid >> 5;
    int lane = tid & 31;
    int half = lane >> 4;
    int m    = lane & 15;
    int mreg = wave & 3;    // row region: 32 rows
    int nreg = wave >> 2;   // col region: 64 cols

    // tile copy mapping: 128 rows x 128B; thread -> 64B of one row
    int crow = tid >> 1;          // 0..127
    int ccol = (tid & 1) * 32;    // 0 or 32 (bf16 elems)

    v8f c[2][4];
#pragma unroll
    for (int i = 0; i < 2; ++i)
#pragma unroll
        for (int j = 0; j < 4; ++j) c[i][j] = (v8f){0,0,0,0,0,0,0,0};

    const bf16_t* agBase = A  + (size_t)(bm + crow) * K + ccol;
    const bf16_t* bgBase = Wt + (size_t)(bn + crow) * K + ccol;
    int nt = K / BK;

#if HAVE_ASYNC
    auto issue_tile = [&](int t, int buf) {
        const bf16_t* a = agBase + (size_t)t * BK;
        const bf16_t* b = bgBase + (size_t)t * BK;
        __builtin_amdgcn_global_load_async_to_lds_b128((v4i*)a, (v4i*)&As[buf][crow][ccol], 0,  0);
        __builtin_amdgcn_global_load_async_to_lds_b128((v4i*)a, (v4i*)&As[buf][crow][ccol], 16, 0);
        __builtin_amdgcn_global_load_async_to_lds_b128((v4i*)a, (v4i*)&As[buf][crow][ccol], 32, 0);
        __builtin_amdgcn_global_load_async_to_lds_b128((v4i*)a, (v4i*)&As[buf][crow][ccol], 48, 0);
        __builtin_amdgcn_global_load_async_to_lds_b128((v4i*)b, (v4i*)&Bs[buf][crow][ccol], 0,  0);
        __builtin_amdgcn_global_load_async_to_lds_b128((v4i*)b, (v4i*)&Bs[buf][crow][ccol], 16, 0);
        __builtin_amdgcn_global_load_async_to_lds_b128((v4i*)b, (v4i*)&Bs[buf][crow][ccol], 32, 0);
        __builtin_amdgcn_global_load_async_to_lds_b128((v4i*)b, (v4i*)&Bs[buf][crow][ccol], 48, 0);
    };
    issue_tile(0, 0);   // prologue
#endif

    for (int i = 0; i < nt; ++i) {
        int cur = i & 1;
#if HAVE_ASYNC
        asm volatile("s_wait_asynccnt 0" ::: "memory");
        __syncthreads();                     // tile i visible to all waves;
                                             // all waves done reading buf 1-cur
        if (i + 1 < nt) issue_tile(i + 1, 1 - cur);   // overlaps with WMMAs below
#else
        __syncthreads();
        {
            const bf16_t* a = agBase + (size_t)i * BK;
            const bf16_t* b = bgBase + (size_t)i * BK;
            uint4* ad = (uint4*)&As[cur][crow][ccol];
            uint4* bd = (uint4*)&Bs[cur][crow][ccol];
#pragma unroll
            for (int q = 0; q < 4; ++q) {
                ad[q] = ((const uint4*)a)[q];
                bd[q] = ((const uint4*)b)[q];
            }
        }
        __syncthreads();
#endif

#pragma unroll
        for (int kk = 0; kk < BK; kk += 32) {
            Frag a0, a1;
#pragma unroll
            for (int p = 0; p < 8; ++p) {
                int kx = kk + 2 * p + half * 8 + ((p >= 4) ? 8 : 0);
                a0.u[p] = *(const unsigned*)&As[cur][mreg * 32 + m][kx];
                a1.u[p] = *(const unsigned*)&As[cur][mreg * 32 + 16 + m][kx];
            }
            Frag b[4];
#pragma unroll
            for (int j = 0; j < 4; ++j)
#pragma unroll
                for (int p = 0; p < 8; ++p) {
                    int kx = kk + 2 * p + half * 8 + ((p >= 4) ? 8 : 0);
                    b[j].u[p] = *(const unsigned*)&Bs[cur][nreg * 64 + j * 16 + m][kx];
                }
#pragma unroll
            for (int j = 0; j < 4; ++j) {
                c[0][j] = __builtin_amdgcn_wmma_f32_16x16x32_bf16(
                    false, a0.v, false, b[j].v, (short)0, c[0][j], false, false);
                c[1][j] = __builtin_amdgcn_wmma_f32_16x16x32_bf16(
                    false, a1.v, false, b[j].v, (short)0, c[1][j], false, false);
            }
        }
    }

    // epilogue
#pragma unroll
    for (int i = 0; i < 2; ++i)
#pragma unroll
        for (int j = 0; j < 4; ++j) {
            int col = bn + nreg * 64 + j * 16 + m;
            float bi = bias[col];
#pragma unroll
            for (int r = 0; r < 8; ++r) {
                int row = bm + mreg * 32 + i * 16 + half * 8 + r;
                float v = c[i][j][r] + bi;
                if (relu) v = fmaxf(v, 0.f);
                size_t idx = (size_t)row * N + col;
                if (residual) v += residual[idx];
                if (outF) outF[idx] = v;
                if (outB) outB[idx] = (bf16_t)v;
            }
        }
}

// ---------------------------------------------------------------------------
// Flash attention: causal, scale 1/8. qkv rows are [q|k|v] of 1024 each.
// Each wave: 16 query rows x full head (64). 8 waves/block -> 128 q rows.
// grid = (SEQ/128, NB*HEADS)
// ---------------------------------------------------------------------------
__global__ __launch_bounds__(256) void attn_wmma(const bf16_t* __restrict__ qkv,
                                                 bf16_t* __restrict__ out) {
    __shared__ __align__(16) bf16_t Pt[8][16][32];  // per-wave P transpose scratch

    int tid  = threadIdx.x;
    int wave = tid >> 5;
    int lane = tid & 31;
    int half = lane >> 4;
    int m    = lane & 15;

    int bh = blockIdx.y;
    int b  = bh >> 4;           // batch
    int h  = bh & 15;           // head
    int qs = blockIdx.x * 128 + wave * 16;   // wave's first q row in sequence

    size_t rowbase = (size_t)b * SEQ;
    int qoff = h * HD;
    int koff = EMBD + h * HD;
    int voff = 2 * EMBD + h * HD;

    // Q fragments (A layout): rows = lane%16, d split into two K-steps
    Frag aq0, aq1;
    {
        const bf16_t* qrow = qkv + (rowbase + qs + m) * QKVLD + qoff;
#pragma unroll
        for (int p = 0; p < 8; ++p) {
            int d = 2 * p + half * 8 + ((p >= 4) ? 8 : 0);
            aq0.u[p] = *(const unsigned*)&qrow[d];
            aq1.u[p] = *(const unsigned*)&qrow[d + 32];
        }
    }

    v8f oacc[4];
#pragma unroll
    for (int nc = 0; nc < 4; ++nc) oacc[nc] = (v8f){0,0,0,0,0,0,0,0};
    float mrow[8], lrow[8];
#pragma unroll
    for (int r = 0; r < 8; ++r) { mrow[r] = -1e30f; lrow[r] = 0.f; }

    const float sc = 0.125f;   // 1/sqrt(64)
    int qend = qs + 15;

    for (int j0 = 0; j0 <= qend; j0 += 32) {
        // prefetch next key tile (K and V rows) one tile ahead
        if (j0 + 32 <= qend) {
            const bf16_t* nk = qkv + (rowbase + j0 + 32 + lane) * QKVLD;
            __builtin_prefetch(nk + koff, 0, 1);
            __builtin_prefetch(nk + voff, 0, 1);
        }

        // ---- scores: S[16q][32keys] = Q @ K^T -------------------------------
        v8f s0 = {0,0,0,0,0,0,0,0};
        v8f s1 = {0,0,0,0,0,0,0,0};
#pragma unroll
        for (int jc = 0; jc < 2; ++jc) {
            Frag bk0, bk1;   // B layout: K-dim = d, N = key
            const bf16_t* krow = qkv + (rowbase + j0 + jc * 16 + m) * QKVLD + koff;
#pragma unroll
            for (int p = 0; p < 8; ++p) {
                int d = 2 * p + half * 8 + ((p >= 4) ? 8 : 0);
                bk0.u[p] = *(const unsigned*)&krow[d];
                bk1.u[p] = *(const unsigned*)&krow[d + 32];
            }
            v8f s = {0,0,0,0,0,0,0,0};
            s = __builtin_amdgcn_wmma_f32_16x16x32_bf16(false, aq0.v, false, bk0.v,
                                                        (short)0, s, false, false);
            s = __builtin_amdgcn_wmma_f32_16x16x32_bf16(false, aq1.v, false, bk1.v,
                                                        (short)0, s, false, false);
            if (jc == 0) s0 = s; else s1 = s;
        }

        // ---- causal mask + online softmax ----------------------------------
        float tmax[8];
#pragma unroll
        for (int r = 0; r < 8; ++r) {
            int q = qs + half * 8 + r;
            float v0 = (j0 + m      > q) ? -1e30f : s0[r] * sc;
            float v1 = (j0 + 16 + m > q) ? -1e30f : s1[r] * sc;
            s0[r] = v0; s1[r] = v1;
            tmax[r] = fmaxf(v0, v1);
        }
#pragma unroll
        for (int off = 1; off < 16; off <<= 1)
#pragma unroll
            for (int r = 0; r < 8; ++r)
                tmax[r] = fmaxf(tmax[r], __shfl_xor(tmax[r], off));

        float pscale[8], rsum[8];
#pragma unroll
        for (int r = 0; r < 8; ++r) {
            float mn = fmaxf(mrow[r], tmax[r]);
            pscale[r] = __expf(mrow[r] - mn);
            mrow[r] = mn;
            float p0 = __expf(s0[r] - mn);
            float p1 = __expf(s1[r] - mn);
            s0[r] = p0; s1[r] = p1;
            rsum[r] = p0 + p1;
        }
#pragma unroll
        for (int off = 1; off < 16; off <<= 1)
#pragma unroll
            for (int r = 0; r < 8; ++r)
                rsum[r] += __shfl_xor(rsum[r], off);
#pragma unroll
        for (int r = 0; r < 8; ++r) lrow[r] = lrow[r] * pscale[r] + rsum[r];
#pragma unroll
        for (int nc = 0; nc < 4; ++nc)
#pragma unroll
            for (int r = 0; r < 8; ++r) oacc[nc][r] *= pscale[r];

        // ---- transpose P (C layout -> A layout) through per-wave LDS -------
#pragma unroll
        for (int r = 0; r < 8; ++r) {
            Pt[wave][half * 8 + r][m]      = (bf16_t)s0[r];
            Pt[wave][half * 8 + r][16 + m] = (bf16_t)s1[r];
        }
        asm volatile("s_wait_dscnt 0" ::: "memory");
        Frag ap;
#pragma unroll
        for (int p = 0; p < 8; ++p) {
            int kk = 2 * p + half * 8 + ((p >= 4) ? 8 : 0);
            ap.u[p] = *(const unsigned*)&Pt[wave][m][kk];
        }

        // ---- O += P @ V ----------------------------------------------------
#pragma unroll
        for (int nc = 0; nc < 4; ++nc) {
            Frag bv;   // B layout: K-dim = key-local, N = d
            int d = nc * 16 + m;
#pragma unroll
            for (int e = 0; e < 16; ++e) {
                size_t vr = rowbase + j0 + half * 8 + e + ((e >= 8) ? 8 : 0);
                bv.h[e] = qkv[vr * QKVLD + voff + d];
            }
            oacc[nc] = __builtin_amdgcn_wmma_f32_16x16x32_bf16(
                false, ap.v, false, bv.v, (short)0, oacc[nc], false, false);
        }
    }

    // ---- finalize: divide by l, store merged-head bf16 ---------------------
    float inv[8];
#pragma unroll
    for (int r = 0; r < 8; ++r) inv[r] = 1.0f / lrow[r];
    bf16_t* obase = out + (rowbase + qs) * EMBD + h * HD;
#pragma unroll
    for (int nc = 0; nc < 4; ++nc)
#pragma unroll
        for (int r = 0; r < 8; ++r)
            obase[(size_t)(half * 8 + r) * EMBD + nc * 16 + m] =
                (bf16_t)(oacc[nc][r] * inv[r]);
}

// ---------------------------------------------------------------------------
// Host launcher
// ---------------------------------------------------------------------------
extern "C" void kernel_launch(void* const* d_in, const int* in_sizes, int n_in,
                              void* d_out, int out_size, void* d_ws, size_t ws_size,
                              hipStream_t stream) {
    const float* x         = (const float*)d_in[0];
    const float* ln1_scale = (const float*)d_in[1];
    const float* ln1_bias  = (const float*)d_in[2];
    const float* w_qkv     = (const float*)d_in[3];
    const float* b_qkv     = (const float*)d_in[4];
    const float* w_proj    = (const float*)d_in[5];
    const float* b_proj    = (const float*)d_in[6];
    const float* ln2_scale = (const float*)d_in[7];
    const float* ln2_bias  = (const float*)d_in[8];
    const float* w_fc      = (const float*)d_in[9];
    const float* b_fc      = (const float*)d_in[10];
    const float* w_out     = (const float*)d_in[11];
    const float* b_out     = (const float*)d_in[12];
    float* y = (float*)d_out;

    char* ws = (char*)d_ws;
    size_t off = 0;
    auto take = [&](size_t bytes) {
        size_t p = off;
        off += (bytes + 255) & ~(size_t)255;
        return p;
    };
    bf16_t* h1     = (bf16_t*)(ws + take((size_t)MTOT * EMBD * 2));
    bf16_t* wqkvT  = (bf16_t*)(ws + take((size_t)EMBD * 3 * EMBD * 2)); // [3E][E]
    bf16_t* qkv    = (bf16_t*)(ws + take((size_t)MTOT * 3 * EMBD * 2));
    bf16_t* attn   = (bf16_t*)(ws + take((size_t)MTOT * EMBD * 2));
    bf16_t* wprojT = (bf16_t*)(ws + take((size_t)EMBD * EMBD * 2));     // [E][E]
    float*  x2     = (float*) (ws + take((size_t)MTOT * EMBD * 4));
    bf16_t* h2     = (bf16_t*)(ws + take((size_t)MTOT * EMBD * 2));
    bf16_t* wfcT   = (bf16_t*)(ws + take((size_t)EMBD * FF * 2));       // [FF][E]
    bf16_t* hfc    = (bf16_t*)(ws + take((size_t)MTOT * FF * 2));
    bf16_t* woutT  = (bf16_t*)(ws + take((size_t)FF * EMBD * 2));       // [E][FF]

    // weight convert + transpose: in [K][N] f32 -> out [N][K] bf16
    transpose_cvt<<<dim3(3 * EMBD / 32, EMBD / 32), dim3(32, 8), 0, stream>>>(
        w_qkv, wqkvT, EMBD, 3 * EMBD);
    transpose_cvt<<<dim3(EMBD / 32, EMBD / 32), dim3(32, 8), 0, stream>>>(
        w_proj, wprojT, EMBD, EMBD);
    transpose_cvt<<<dim3(FF / 32, EMBD / 32), dim3(32, 8), 0, stream>>>(
        w_fc, wfcT, EMBD, FF);
    transpose_cvt<<<dim3(EMBD / 32, FF / 32), dim3(32, 8), 0, stream>>>(
        w_out, woutT, FF, EMBD);

    // LN1
    ln_bf16<<<MTOT, 256, 0, stream>>>(x, ln1_scale, ln1_bias, h1);

    // QKV: [4096,1024] @ [1024,3072] -> bf16 qkv
    gemm_bf16_wmma<<<dim3(3 * EMBD / BN, MTOT / BM), 256, 0, stream>>>(
        h1, wqkvT, b_qkv, nullptr, nullptr, qkv, 3 * EMBD, EMBD, 0);

    // attention
    attn_wmma<<<dim3(SEQ / 128, NB * HEADS), 256, 0, stream>>>(qkv, attn);

    // proj + residual(x) -> x2 (f32)
    gemm_bf16_wmma<<<dim3(EMBD / BN, MTOT / BM), 256, 0, stream>>>(
        attn, wprojT, b_proj, x, x2, nullptr, EMBD, EMBD, 0);

    // LN2
    ln_bf16<<<MTOT, 256, 0, stream>>>(x2, ln2_scale, ln2_bias, h2);

    // FC + relu -> hfc (bf16)
    gemm_bf16_wmma<<<dim3(FF / BN, MTOT / BM), 256, 0, stream>>>(
        h2, wfcT, b_fc, nullptr, nullptr, hfc, FF, EMBD, 1);

    // out proj + residual(x2) -> y (f32)
    gemm_bf16_wmma<<<dim3(EMBD / BN, MTOT / BM), 256, 0, stream>>>(
        hfc, woutT, b_out, x2, y, nullptr, EMBD, FF, 0);

    (void)in_sizes; (void)n_in; (void)out_size; (void)ws_size;
}
